// GATv2Model_2783138808355
// MI455X (gfx1250) — matrix operand order repaired and checked by reading, not compiled, not run on hardware
//
#include <hip/hip_runtime.h>
#include <hip/hip_bf16.h>

// ---------------------------------------------------------------------------
// GATv2 (3 layers) for MI455X / gfx1250, wave32.
//   - Dense projections: f16 WMMA (v_wmma_f32_16x16x32_f16), f32 accumulate.
//   - Edge passes: one wave per edge, shfl_xor head reductions, float atomics.
// ---------------------------------------------------------------------------

typedef __attribute__((ext_vector_type(8)))  _Float16 v8h;
typedef __attribute__((ext_vector_type(16))) _Float16 v16h;
typedef __attribute__((ext_vector_type(8)))  float    v8f;

#define NEG_SLOPE 0.2f

// ------------------------------ small utils -------------------------------

__global__ void fill_f32_kernel(float* p, float v, int n) {
    int t = blockIdx.x * blockDim.x + threadIdx.x;
    if (t < n) p[t] = v;
}

__global__ void fill_u32_kernel(unsigned* p, unsigned v, int n) {
    int t = blockIdx.x * blockDim.x + threadIdx.x;
    if (t < n) p[t] = v;
}

// x fp32 -> f16 (GEMM A operand)
__global__ void cvt_f16_kernel(const float* __restrict__ x, _Float16* __restrict__ xh, int n) {
    int t = blockIdx.x * blockDim.x + threadIdx.x;
    if (t < n) xh[t] = (_Float16)x[t];
}

// W [K x ncols] fp32 row-major  ->  WT [ncols x K] f16 row-major
__global__ void wt_transpose_kernel(const float* __restrict__ W, _Float16* __restrict__ WT,
                                    int K, int ncols) {
    int t = blockIdx.x * blockDim.x + threadIdx.x;
    if (t >= K * ncols) return;
    int n = t / K, k = t % K;
    WT[n * K + k] = (_Float16)W[k * ncols + n];
}

// ------------------------------ WMMA GEMM ---------------------------------
// C[M x ncols] = A[M x K](f16) * B[K x ncols](f16, given transposed) + bias
// One wave per 16x16 output tile; M = 50000 is an exact multiple of 16.
__global__ void __launch_bounds__(256)
gemm_wmma_f16_kernel(const _Float16* __restrict__ A, const _Float16* __restrict__ BT,
                     const float* __restrict__ bias, float* __restrict__ Cout,
                     int K, int ncols) {
    const int lane    = threadIdx.x & 31;
    const int wave    = threadIdx.x >> 5;
    const int rowBase = blockIdx.x * 16;
    const int colBase = wave * 16;

    // A fragment addressing (16x32 f16, wave32 layout):
    //   lanes 0-15 : row = lane,   K in {k0+0..7} U {k0+16..23}
    //   lanes 16-31: row = lane-16,K in {k0+8..15} U {k0+24..31}
    const int rA     = rowBase + (lane & 15);
    const int khalfA = (lane >> 4) * 8;
    // B fragment addressing (32x16 f16): lane holds column n, K in same order as A
    const int nB     = colBase + (lane & 15);
    const int khalfB = (lane >> 4) * 8;

    v8f acc = {};
    for (int k0 = 0; k0 < K; k0 += 32) {
        union { v16h v; v8h h[2]; } a, b;
        const _Float16* ap = A  + (size_t)rA * K + k0;
        const _Float16* bp = BT + (size_t)nB * K + k0;
        a.h[0] = *(const v8h*)(ap + khalfA);
        a.h[1] = *(const v8h*)(ap + 16 + khalfA);
        b.h[0] = *(const v8h*)(bp + khalfB);
        b.h[1] = *(const v8h*)(bp + 16 + khalfB);
        acc = __builtin_amdgcn_wmma_f32_16x16x32_f16(
            /*neg_a=*/false, a.v, /*neg_b=*/false, b.v,
            /*c_mod=*/(short)0, acc, /*reuse_a=*/false, /*reuse_b=*/false);
    }

    // C/D layout: VGPR r -> row r (lanes 0-15) / row 8+r (lanes 16-31), col = lane%16
    const float bv     = bias[nB];
    const int   rowOff = (lane >> 4) * 8;
#pragma unroll
    for (int r = 0; r < 8; ++r)
        Cout[(size_t)(rowBase + rowOff + r) * ncols + nB] = acc[r] + bv;
}

// --------------------------- edge pass kernels ----------------------------

__device__ __forceinline__ unsigned f32_key(float f) {
    unsigned b = __float_as_uint(f);
    return (b & 0x80000000u) ? ~b : (b | 0x80000000u);
}
__device__ __forceinline__ float key_f32(unsigned k) {
    unsigned b = (k & 0x80000000u) ? (k & 0x7FFFFFFFu) : ~k;
    return __uint_as_float(b);
}

// Pass A: logit[e][h] = att[h] . leaky(xl[src] + xr[dst]); atomic segment-max.
template <int H, int C>
__global__ void edge_logits_kernel(const float* __restrict__ xl, const float* __restrict__ xr,
                                   const int* __restrict__ src, const int* __restrict__ dst,
                                   const float* __restrict__ att,
                                   float* __restrict__ logit, unsigned* __restrict__ mkey,
                                   int E) {
    constexpr int HC  = H * C;
    constexpr int LPH = 32 / H;   // lanes per head
    constexpr int CPL = HC / 32;  // channels per lane
    const int wid  = (blockIdx.x * blockDim.x + threadIdx.x) >> 5;
    const int lane = threadIdx.x & 31;
    if (wid >= E) return;
    const int s = src[wid], d = dst[wid];
    const int h     = lane / LPH;
    const int cin   = (lane % LPH) * CPL;      // channel within head
    const int cbase = h * C + cin;             // channel within HC
    float partial = 0.f;
#pragma unroll
    for (int j = 0; j < CPL; ++j) {
        float e = xl[(size_t)s * HC + cbase + j] + xr[(size_t)d * HC + cbase + j];
        e = e > 0.f ? e : NEG_SLOPE * e;
        partial += att[h * C + cin + j] * e;
    }
#pragma unroll
    for (int off = LPH >> 1; off >= 1; off >>= 1)
        partial += __shfl_xor(partial, off, 32);
    if ((lane % LPH) == 0) {
        logit[(size_t)wid * H + h] = partial;
        atomicMax(&mkey[d * H + h], f32_key(partial));
    }
}

// Pass B: denom[dst][h] += exp(logit - m[dst][h])
__global__ void edge_denom_kernel(const float* __restrict__ logit, const unsigned* __restrict__ mkey,
                                  const int* __restrict__ dst, float* __restrict__ denom,
                                  int E, int H) {
    int t = blockIdx.x * blockDim.x + threadIdx.x;
    if (t >= E * H) return;
    int e = t / H, h = t - e * H;
    int d = dst[e];
    float m = key_f32(mkey[d * H + h]);
    atomicAdd(&denom[d * H + h], __expf(logit[t] - m));
}

// Pass C: acc[dst][h][c] += xl[src][h][c] * alpha[e][h]
template <int H, int C>
__global__ void edge_scatter_kernel(const float* __restrict__ xl,
                                    const float* __restrict__ logit,
                                    const unsigned* __restrict__ mkey,
                                    const float* __restrict__ denom,
                                    const int* __restrict__ src, const int* __restrict__ dst,
                                    float* __restrict__ acc, int E) {
    constexpr int HC  = H * C;
    constexpr int LPH = 32 / H;
    constexpr int CPL = HC / 32;
    const int wid  = (blockIdx.x * blockDim.x + threadIdx.x) >> 5;
    const int lane = threadIdx.x & 31;
    if (wid >= E) return;
    const int s = src[wid], d = dst[wid];
    const int h     = lane / LPH;
    const int cbase = h * C + (lane % LPH) * CPL;
    const float m     = key_f32(mkey[d * H + h]);
    const float alpha = __expf(logit[(size_t)wid * H + h] - m) / denom[d * H + h];
#pragma unroll
    for (int j = 0; j < CPL; ++j)
        atomicAdd(&acc[(size_t)d * HC + cbase + j],
                  xl[(size_t)s * HC + cbase + j] * alpha);
}

// Epilogue: h = (elu?)(acc + bias); also emit f16 copy for the next GEMM.
__global__ void bias_act_kernel(const float* __restrict__ acc, const float* __restrict__ bias,
                                float* __restrict__ hout, _Float16* __restrict__ hh,
                                int n, int HC, int doElu) {
    int t = blockIdx.x * blockDim.x + threadIdx.x;
    if (t >= n) return;
    float v = acc[t] + bias[t % HC];
    if (doElu) v = v > 0.f ? v : (__expf(v) - 1.f);
    hout[t] = v;
    if (hh) hh[t] = (_Float16)v;
}

// --------------------------- global mean pool -----------------------------

__global__ void pool_acc_kernel(const float* __restrict__ hfin, const int* __restrict__ batch,
                                float* __restrict__ sums, float* __restrict__ counts, int N) {
    int t = blockIdx.x * blockDim.x + threadIdx.x;
    if (t >= N * 64) return;
    int n = t >> 6, c = t & 63;
    int g = batch[n];
    atomicAdd(&sums[g * 64 + c], hfin[(size_t)n * 64 + c]);
    if (c == 0) atomicAdd(&counts[g], 1.f);
}

__global__ void pool_div_kernel(const float* __restrict__ sums, const float* __restrict__ counts,
                                float* __restrict__ out, int G) {
    int t = blockIdx.x * blockDim.x + threadIdx.x;
    if (t >= G * 64) return;
    out[t] = sums[t] / fmaxf(counts[t >> 6], 1.0f);
}

// ------------------------------- host side --------------------------------

static inline size_t align256(size_t x) { return (x + 255) & ~(size_t)255; }

extern "C" void kernel_launch(void* const* d_in, const int* in_sizes, int n_in,
                              void* d_out, int out_size, void* d_ws, size_t ws_size,
                              hipStream_t stream) {
    (void)n_in; (void)ws_size;
    const int IN_CH = 128, HID = 128;
    const int N = in_sizes[0] / IN_CH;      // 50000
    const int E = in_sizes[1] / 2;          // 800000
    const int G = out_size / 64;            // 64 graphs

    const float* x    = (const float*)d_in[0];
    const int*   eidx = (const int*)d_in[1];
    const int*   src  = eidx;
    const int*   dst  = eidx + E;
    const int*   batch = (const int*)d_in[2];

    // per-layer params: Wl, bl, Wr, br, att, bias at d_in[3 + 6*l ...]
    const float* Wl[3];  const float* bl[3];
    const float* Wr[3];  const float* br[3];
    const float* att[3]; const float* bias[3];
    for (int l = 0; l < 3; ++l) {
        Wl[l]   = (const float*)d_in[3 + 6 * l + 0];
        bl[l]   = (const float*)d_in[3 + 6 * l + 1];
        Wr[l]   = (const float*)d_in[3 + 6 * l + 2];
        br[l]   = (const float*)d_in[3 + 6 * l + 3];
        att[l]  = (const float*)d_in[3 + 6 * l + 4];
        bias[l] = (const float*)d_in[3 + 6 * l + 5];
    }
    const int HC[3] = {128, 128, 64};
    const int Hh[3] = {4, 4, 1};

    // ---- workspace layout ----
    char* ws = (char*)d_ws;
    size_t off = 0;
    auto take = [&](size_t bytes) { char* p = ws + off; off += align256(bytes); return p; };
    _Float16* xh    = (_Float16*)take((size_t)N * 128 * sizeof(_Float16));
    float*    xlb   = (float*)   take((size_t)N * 128 * sizeof(float));
    float*    xrb   = (float*)   take((size_t)N * 128 * sizeof(float));
    float*    accb  = (float*)   take((size_t)N * 128 * sizeof(float));
    float*    hbuf  = (float*)   take((size_t)N * 128 * sizeof(float));
    _Float16* wtl   = (_Float16*)take((size_t)128 * 128 * sizeof(_Float16));
    _Float16* wtr   = (_Float16*)take((size_t)128 * 128 * sizeof(_Float16));
    float*    logit = (float*)   take((size_t)E * 4 * sizeof(float));
    unsigned* mkey  = (unsigned*)take((size_t)N * 4 * sizeof(unsigned));
    float*    denom = (float*)   take((size_t)N * 4 * sizeof(float));
    float*    sums  = (float*)   take((size_t)G * 64 * sizeof(float));
    float*    cnts  = (float*)   take((size_t)G * sizeof(float));

    const int TB = 256;
    auto blocks = [](long long n, int tb) { return (unsigned)((n + tb - 1) / tb); };

    // x -> f16 for layer 0
    cvt_f16_kernel<<<blocks((long long)N * 128, TB), TB, 0, stream>>>(x, xh, N * 128);

    for (int l = 0; l < 3; ++l) {
        const int hc = HC[l], H = Hh[l], C = hc / H, K = 128;

        // weight prep (transpose + f16)
        wt_transpose_kernel<<<blocks(K * hc, TB), TB, 0, stream>>>(Wl[l], wtl, K, hc);
        wt_transpose_kernel<<<blocks(K * hc, TB), TB, 0, stream>>>(Wr[l], wtr, K, hc);

        // projections via WMMA: xl = xh@Wl + bl ; xr = xh@Wr + br
        dim3 gblk((unsigned)(hc / 16) * 32);
        dim3 ggrd((unsigned)(N / 16));
        gemm_wmma_f16_kernel<<<ggrd, gblk, 0, stream>>>(xh, wtl, bl[l], xlb, K, hc);
        gemm_wmma_f16_kernel<<<ggrd, gblk, 0, stream>>>(xh, wtr, br[l], xrb, K, hc);

        // reset segment-reduction state
        fill_u32_kernel<<<blocks(N * H, TB), TB, 0, stream>>>(mkey, 0u, N * H);
        fill_f32_kernel<<<blocks(N * H, TB), TB, 0, stream>>>(denom, 0.f, N * H);
        fill_f32_kernel<<<blocks((long long)N * hc, TB), TB, 0, stream>>>(accb, 0.f, N * hc);

        // edge passes: one wave per edge
        unsigned eb = blocks((long long)E * 32, TB);
        if (H == 4) {
            edge_logits_kernel<4, 32><<<eb, TB, 0, stream>>>(xlb, xrb, src, dst, att[l], logit, mkey, E);
        } else {
            edge_logits_kernel<1, 64><<<eb, TB, 0, stream>>>(xlb, xrb, src, dst, att[l], logit, mkey, E);
        }
        edge_denom_kernel<<<blocks((long long)E * H, TB), TB, 0, stream>>>(logit, mkey, dst, denom, E, H);
        if (H == 4) {
            edge_scatter_kernel<4, 32><<<eb, TB, 0, stream>>>(xlb, logit, mkey, denom, src, dst, accb, E);
        } else {
            edge_scatter_kernel<1, 64><<<eb, TB, 0, stream>>>(xlb, logit, mkey, denom, src, dst, accb, E);
        }

        // epilogue: bias (+ELU for layers 0/1), emit fp32 h and f16 copy for next GEMM
        int doElu = (l < 2) ? 1 : 0;
        bias_act_kernel<<<blocks((long long)N * hc, TB), TB, 0, stream>>>(
            accb, bias[l], hbuf, (l < 2) ? xh : (_Float16*)nullptr, N * hc, hc, doElu);
    }

    // global mean pool over graphs
    fill_f32_kernel<<<blocks(G * 64, TB), TB, 0, stream>>>(sums, 0.f, G * 64);
    fill_f32_kernel<<<blocks(G, TB), TB, 0, stream>>>(cnts, 0.f, G);
    pool_acc_kernel<<<blocks((long long)N * 64, TB), TB, 0, stream>>>(hbuf, batch, sums, cnts, N);
    pool_div_kernel<<<blocks(G * 64, TB), TB, 0, stream>>>(sums, cnts, (float*)d_out, G);
}
